// MultilayersGAT_79199196938542
// MI455X (gfx1250) — matrix operand order repaired and checked by reading, not compiled, not run on hardware
//
#include <hip/hip_runtime.h>
#include <hip/hip_bf16.h>

typedef __attribute__((ext_vector_type(16))) _Float16 v16h;
typedef __attribute__((ext_vector_type(8)))  float    v8f;

namespace gat {
constexpr int N = 1024, FIN = 128, ED = 16, H = 4, F = 32, HF = 128, FOUT = 128, NL = 3;
constexpr int OUTC = FIN + NL * FOUT;      // 512 output columns
constexpr float ALPHA = 0.2f;
constexpr float NEGINF = -9.0e15f;
constexpr int MT   = N / 16;               // 64 row tiles
constexpr int KT3  = N / 32;               // 32 k-tiles for aggregation GEMM
constexpr int FRAG = 512;                  // halves per 16x32 (A) or 32x16 (B) fragment

// ---------------- workspace layout (bytes) ----------------
constexpr size_t ALN = 256;
constexpr size_t aup(size_t x) { return (x + ALN - 1) & ~(ALN - 1); }
constexpr size_t SZ_ATT = (size_t)H * MT * KT3 * FRAG * sizeof(_Float16); // 8 MB
constexpr size_t SZ_HQF = (size_t)H * 2 * KT3 * FRAG * sizeof(_Float16);  // 256 KB
constexpr size_t SZ_WF  = (size_t)4 * 8 * FRAG * sizeof(_Float16);        // 32 KB
constexpr size_t OFF_ATTG = 0;
constexpr size_t OFF_ATTN = aup(OFF_ATTG + SZ_ATT);
constexpr size_t OFF_HQF  = aup(OFF_ATTN + SZ_ATT);
constexpr size_t OFF_HQEF = aup(OFF_HQF  + SZ_HQF);
constexpr size_t OFF_WQF  = aup(OFF_HQEF + SZ_HQF);
constexpr size_t OFF_WQEF = aup(OFF_WQF  + SZ_WF);
constexpr size_t OFF_WOF  = aup(OFF_WQEF + SZ_WF);
constexpr size_t OFF_U    = aup(OFF_WOF  + SZ_WF);                 // 128 x 12 f32
constexpr size_t OFF_V    = aup(OFF_U    + (size_t)FIN * 12 * 4);  // 4 x 16 f32
constexpr size_t OFF_SQ   = aup(OFF_V    + (size_t)H * ED * 4);    // N x 12 f32
constexpr size_t OFF_EW   = aup(OFF_SQ   + (size_t)N * 12 * 4);    // N x H x 16 f32
constexpr size_t OFF_XCAT = aup(OFF_EW   + (size_t)N * H * ED * 4);
constexpr size_t OFF_XNXT = aup(OFF_XCAT + (size_t)N * HF * 4);
} // namespace gat

using namespace gat;

// ---------------- fragment index helpers (contiguous k order: k = 16*hs + slot) ----------------
// A fragment (16x32 f16): lane = (M&15) + 16*hs ; within v16h:
//   slot s <-> K = 16*hs + s
__device__ __forceinline__ int afrag_idx(int m, int k) {
  int hs   = k >> 4;
  int lane = m + (hs << 4);
  int slot = k & 15;
  return lane * 16 + slot;
}
// B fragment (32x16 f16): lane = N + 16*(K>>4); slot = K&15
__device__ __forceinline__ int bfrag_idx(int k, int n) {
  return (n + ((k >> 4) << 4)) * 16 + (k & 15);
}

__device__ __forceinline__ v8f wmma_f16(v16h a, v16h b, v8f c) {
  return __builtin_amdgcn_wmma_f32_16x16x32_f16(false, a, false, b, (short)0, c,
                                                false, false);
}

// build an A fragment (rows of a row-major f32 matrix, converted to f16)
__device__ __forceinline__ v16h load_afrag_f32(const float* X, int ld, int row0,
                                               int col0, int lane) {
  int m = lane & 15, hs = lane >> 4;
  v16h a;
#pragma unroll
  for (int r = 0; r < 8; ++r) {
    int kb = (hs << 4) + (r << 1);
    float2 xv = *(const float2*)(X + (size_t)(row0 + m) * ld + col0 + kb);
    a[2 * r]     = (_Float16)xv.x;
    a[2 * r + 1] = (_Float16)xv.y;
  }
  return a;
}

// ======================= Kcopy: d_out[:, :128] = x0 =======================
__global__ __launch_bounds__(256) void k_copy_x(const float* __restrict__ x,
                                                float* __restrict__ out) {
  int t = blockIdx.x * 256 + threadIdx.x;
  out[(size_t)(t >> 7) * OUTC + (t & 127)] = x[t];
}

// ======================= P0: per-layer weight prep (grid-strided) ============
// U (128x12): cols 0-3 = Wk_h@a1, 4-7 = Wq_h@a2, 8-11 = Wqe1_h@a2
// V (4x16):   v_h = Wqe2_h @ a2
// wqf/wqef/wof: B-fragment-packed f16 weight matrices [kt(4)][w(8)][512]
__global__ __launch_bounds__(256) void k_prep(const float* __restrict__ WkL,
                                              const float* __restrict__ WqL,
                                              const float* __restrict__ WqeL,
                                              const float* __restrict__ WoutL,
                                              const float* __restrict__ aL,
                                              float* __restrict__ U,
                                              float* __restrict__ V,
                                              _Float16* __restrict__ wqf,
                                              _Float16* __restrict__ wqef,
                                              _Float16* __restrict__ wof) {
  int gt = blockIdx.x * 256 + threadIdx.x;
  int gs = gridDim.x * 256;
  // U
  for (int t = gt; t < FIN * 12; t += gs) {
    int c = t / 12, col = t % 12, ty = col >> 2, h = col & 3;
    float s = 0.f;
    if (ty == 0) {
      for (int f = 0; f < F; ++f) s += WkL[((size_t)(h * FIN + c)) * F + f] * aL[h * 2 * F + f];
    } else if (ty == 1) {
      for (int f = 0; f < F; ++f) s += WqL[((size_t)(h * FIN + c)) * F + f] * aL[h * 2 * F + F + f];
    } else {
      for (int f = 0; f < F; ++f) s += WqeL[((size_t)(h * (FIN + ED) + c)) * F + f] * aL[h * 2 * F + F + f];
    }
    U[c * 12 + col] = s;
  }
  // V
  for (int t = gt; t < H * ED; t += gs) {
    int h = t >> 4, c = t & 15;
    float s = 0.f;
    for (int f = 0; f < F; ++f)
      s += WqeL[((size_t)(h * (FIN + ED) + FIN + c)) * F + f] * aL[h * 2 * F + F + f];
    V[h * ED + c] = s;
  }
  // fragment packs (16384 halves each): idx = ((kt*8 + w)*512) + lane*16 + slot
  for (int t = gt; t < 4 * 8 * FRAG; t += gs) {
    int kt = t >> 12, rem = t & 4095, w = rem >> 9, s = rem & 511;
    int lane = s >> 4, slot = s & 15;
    int kl = slot + ((lane >> 4) << 4);   // 0..31
    int n  = lane & 15;
    int k  = kt * 32 + kl;                // row (0..127)
    int col = w * 16 + n;                 // col (0..127)
    int h = col >> 5, f = col & 31;
    wqf[t]  = (_Float16)WqL[((size_t)(h * FIN + k)) * F + f];
    wqef[t] = (_Float16)WqeL[((size_t)(h * (FIN + ED) + k)) * F + f];
    wof[t]  = (_Float16)WoutL[(size_t)k * FOUT + col];
  }
}

// ======================= K1: projections (WMMA) =======================
// HQ = x@Wq_cat, HQE = x@Wqe1_cat, results written in K3 B-fragment layout.
__global__ __launch_bounds__(256) void k_proj(const float* __restrict__ X,
                                              const _Float16* __restrict__ wqf,
                                              const _Float16* __restrict__ wqef,
                                              _Float16* __restrict__ hqf,
                                              _Float16* __restrict__ hqef) {
  int mt = blockIdx.x;
  int wv = threadIdx.x >> 5, lane = threadIdx.x & 31;
  v8f accq = {}, accqe = {};
#pragma unroll
  for (int kt = 0; kt < 4; ++kt) {
    v16h a  = load_afrag_f32(X, FIN, mt * 16, kt * 32, lane);
    v16h bq = *(const v16h*)(wqf  + ((size_t)(kt * 8 + wv)) * FRAG + lane * 16);
    v16h be = *(const v16h*)(wqef + ((size_t)(kt * 8 + wv)) * FRAG + lane * 16);
    accq  = wmma_f16(a, bq, accq);
    accqe = wmma_f16(a, be, accqe);
  }
#pragma unroll
  for (int r = 0; r < 8; ++r) {
    int m = r + ((lane >> 4) << 3);
    int row = mt * 16 + m;
    int col = wv * 16 + (lane & 15);
    size_t base = ((size_t)((col >> 5) * 2 + ((col >> 4) & 1)) * KT3 + (row >> 5)) * FRAG;
    int fi = bfrag_idx(row & 31, col & 15);
    hqf[base + fi]  = (_Float16)accq[r];
    hqef[base + fi] = (_Float16)accqe[r];
  }
}

// ======================= K1b: collapsed score scalars =======================
__global__ __launch_bounds__(256) void k_sq(const float* __restrict__ X,
                                            const float* __restrict__ U,
                                            float* __restrict__ SQ) {
  int i = blockIdx.x * 256 + threadIdx.x;
  const float4* X4 = (const float4*)(X + (size_t)i * FIN);
  float s[12];
#pragma unroll
  for (int c = 0; c < 12; ++c) s[c] = 0.f;
  for (int c4 = 0; c4 < FIN / 4; ++c4) {
    float4 xv = X4[c4];
    float xs[4] = {xv.x, xv.y, xv.z, xv.w};
#pragma unroll
    for (int q = 0; q < 4; ++q) {
#pragma unroll
      for (int col = 0; col < 12; ++col) s[col] += xs[q] * U[(c4 * 4 + q) * 12 + col];
    }
  }
#pragma unroll
  for (int col = 0; col < 12; ++col) SQ[i * 12 + col] = s[col];
}

// ======================= K2: scores + softmax + packed attention =======================
__global__ __launch_bounds__(256) void k_attn(const float* __restrict__ E,
                                              const int* __restrict__ adj,
                                              const float* __restrict__ SQ,
                                              const float* __restrict__ V,
                                              _Float16* __restrict__ attg,
                                              _Float16* __restrict__ attng,
                                              float* __restrict__ EW) {
  __shared__ float sc[H][N];
  __shared__ unsigned char flg[N];
  __shared__ float red[256];
  __shared__ float vls[H][ED];
  __shared__ float skl[H];
  int i = blockIdx.x, t = threadIdx.x;
  if (t < H * ED) vls[t >> 4][t & 15] = V[t];
  if (t < H) skl[t] = SQ[i * 12 + t];
  __syncthreads();

  // pass A: scores for all heads (e row streamed as b128 loads)
  for (int j = t; j < N; j += 256) {
    int av = adj[(size_t)i * N + j];
    flg[j] = (unsigned char)((av > 0 ? 1 : 0) | (av > 1 ? 2 : 0));
    const float4* ep4 = (const float4*)(E + ((size_t)i * N + j) * ED);
    __builtin_prefetch((const float*)(ep4) + 256 * ED, 0, 1);  // prefetch ahead
    float4 e4[4];
#pragma unroll
    for (int q = 0; q < 4; ++q) e4[q] = ep4[q];
    const float* ev = (const float*)e4;
#pragma unroll
    for (int h = 0; h < H; ++h) {
      float es = 0.f;
#pragma unroll
      for (int c = 0; c < ED; ++c) es += ev[c] * vls[h][c];
      float hqd = (av > 1) ? (SQ[j * 12 + 8 + h] + es) : SQ[j * 12 + 4 + h];
      float s = skl[h] + hqd;
      s = s > 0.f ? s : ALPHA * s;
      sc[h][j] = (av > 0) ? s : NEGINF;
    }
  }
  __syncthreads();

  // pass B: per-head max / sum-exp reductions
  float mh[H], sh[H];
  for (int h = 0; h < H; ++h) {
    float m = -3.4e38f;
    for (int j = t; j < N; j += 256) m = fmaxf(m, sc[h][j]);
    red[t] = m; __syncthreads();
    for (int off = 128; off > 0; off >>= 1) {
      if (t < off) red[t] = fmaxf(red[t], red[t + off]);
      __syncthreads();
    }
    mh[h] = red[0]; __syncthreads();
    float s = 0.f;
    for (int j = t; j < N; j += 256) s += __expf(sc[h][j] - mh[h]);
    red[t] = s; __syncthreads();
    for (int off = 128; off > 0; off >>= 1) {
      if (t < off) red[t] += red[t + off];
      __syncthreads();
    }
    sh[h] = red[0]; __syncthreads();
  }

  // pass C: write att*(gate)/att*(1-gate) in A-fragment layout, accumulate EW
  int g = t >> 6, tl = t & 63;
  float mg = mh[g], rs = 1.f / sh[g];
  float ew[ED];
#pragma unroll
  for (int c = 0; c < ED; ++c) ew[c] = 0.f;
  for (int j = tl; j < N; j += 64) {
    float att = __expf(sc[g][j] - mg) * rs;
    int fl = flg[j];
    float ag  = (fl & 2) ? att : 0.f;
    float ang = att - ag;
    size_t fb = (((size_t)(g * MT + (i >> 4))) * KT3 + (j >> 5)) * FRAG +
                afrag_idx(i & 15, j & 31);
    attg[fb]  = (_Float16)ag;
    attng[fb] = (_Float16)ang;
    if (fl & 2) {
      const float4* ep4 = (const float4*)(E + ((size_t)i * N + j) * ED);  // L2 hit
#pragma unroll
      for (int q = 0; q < 4; ++q) {
        float4 e4 = ep4[q];
        ew[4 * q + 0] += att * e4.x;
        ew[4 * q + 1] += att * e4.y;
        ew[4 * q + 2] += att * e4.z;
        ew[4 * q + 3] += att * e4.w;
      }
    }
  }
  __syncthreads();              // all sc reads done; reuse sc as reduction scratch
  float* rz = &sc[0][0];
#pragma unroll
  for (int c = 0; c < ED; ++c) rz[t * ED + c] = ew[c];
  __syncthreads();
  if (t < 64) {
    int g2 = t >> 4, c = t & 15;
    float s = 0.f;
    for (int u = 0; u < 64; ++u) s += rz[(g2 * 64 + u) * ED + c];
    EW[((size_t)i * H + g2) * ED + c] = s;
  }
}

// ======================= K3: attention aggregation (WMMA) =======================
__global__ __launch_bounds__(64) void k_aggr(const _Float16* __restrict__ attg,
                                             const _Float16* __restrict__ attng,
                                             const _Float16* __restrict__ hqf,
                                             const _Float16* __restrict__ hqef,
                                             const float* __restrict__ EW,
                                             const float* __restrict__ WqeL,
                                             float* __restrict__ xcat,
                                             float* __restrict__ xnext) {
  __shared__ float wqe2[ED][F];
  __shared__ float ewl[16][ED];
  int mt = blockIdx.x, h = blockIdx.y;
  int wv = threadIdx.x >> 5, lane = threadIdx.x & 31;
  for (int t = threadIdx.x; t < ED * F; t += 64) {
    int c = t >> 5, f = t & 31;
    wqe2[c][f] = WqeL[((size_t)(h * (FIN + ED) + FIN + c)) * F + f];
  }
  for (int t = threadIdx.x; t < 16 * ED; t += 64) {
    int m = t >> 4, c = t & 15;
    ewl[m][c] = EW[((size_t)(mt * 16 + m) * H + h) * ED + c];
  }
  __syncthreads();

  const _Float16* ag = attg  + ((size_t)(h * MT + mt)) * KT3 * FRAG;
  const _Float16* an = attng + ((size_t)(h * MT + mt)) * KT3 * FRAG;
  const _Float16* bq = hqf  + ((size_t)(h * 2 + wv)) * KT3 * FRAG;
  const _Float16* be = hqef + ((size_t)(h * 2 + wv)) * KT3 * FRAG;
  v8f acc = {};
  for (int kt = 0; kt < KT3; ++kt) {
    v16h a1 = *(const v16h*)(an + (size_t)kt * FRAG + lane * 16);
    v16h b1 = *(const v16h*)(bq + (size_t)kt * FRAG + lane * 16);
    v16h a2 = *(const v16h*)(ag + (size_t)kt * FRAG + lane * 16);
    v16h b2 = *(const v16h*)(be + (size_t)kt * FRAG + lane * 16);
    acc = wmma_f16(a1, b1, acc);
    acc = wmma_f16(a2, b2, acc);
  }
#pragma unroll
  for (int r = 0; r < 8; ++r) {
    int m = r + ((lane >> 4) << 3);
    int row = mt * 16 + m;
    int f = wv * 16 + (lane & 15);
    int col = h * F + f;
    float val = acc[r];
#pragma unroll
    for (int c = 0; c < ED; ++c) val += ewl[m][c] * wqe2[c][f];
    xcat[(size_t)row * HF + col]  = val;
    xnext[(size_t)row * HF + col] = val > 0.f ? val : (__expf(val) - 1.f);
  }
}

// ======================= K4: output projection (WMMA) + elu =======================
__global__ __launch_bounds__(256) void k_outp(const float* __restrict__ xcat,
                                              const _Float16* __restrict__ wof,
                                              float* __restrict__ out, int l) {
  int mt = blockIdx.x;
  int wv = threadIdx.x >> 5, lane = threadIdx.x & 31;
  v8f acc = {};
#pragma unroll
  for (int kt = 0; kt < 4; ++kt) {
    v16h a = load_afrag_f32(xcat, HF, mt * 16, kt * 32, lane);
    v16h b = *(const v16h*)(wof + ((size_t)(kt * 8 + wv)) * FRAG + lane * 16);
    acc = wmma_f16(a, b, acc);
  }
#pragma unroll
  for (int r = 0; r < 8; ++r) {
    int m = r + ((lane >> 4) << 3);
    int row = mt * 16 + m;
    int col = wv * 16 + (lane & 15);
    float v = acc[r];
    v = v > 0.f ? v : (__expf(v) - 1.f);
    out[(size_t)row * OUTC + FIN + l * FOUT + col] = v;
  }
}

// ======================= host =======================
extern "C" void kernel_launch(void* const* d_in, const int* in_sizes, int n_in,
                              void* d_out, int out_size, void* d_ws, size_t ws_size,
                              hipStream_t stream) {
  (void)in_sizes; (void)n_in; (void)out_size; (void)ws_size;
  const float* x    = (const float*)d_in[0];
  const float* e    = (const float*)d_in[1];
  const float* Wk   = (const float*)d_in[2];
  const float* Wqe  = (const float*)d_in[3];
  const float* Wq   = (const float*)d_in[4];
  const float* a    = (const float*)d_in[5];
  const float* Wout = (const float*)d_in[6];
  const int*   adj  = (const int*)d_in[7];
  float* out = (float*)d_out;
  char* ws = (char*)d_ws;

  _Float16* attg  = (_Float16*)(ws + OFF_ATTG);
  _Float16* attng = (_Float16*)(ws + OFF_ATTN);
  _Float16* hqf   = (_Float16*)(ws + OFF_HQF);
  _Float16* hqef  = (_Float16*)(ws + OFF_HQEF);
  _Float16* wqf   = (_Float16*)(ws + OFF_WQF);
  _Float16* wqef  = (_Float16*)(ws + OFF_WQEF);
  _Float16* wof   = (_Float16*)(ws + OFF_WOF);
  float* U    = (float*)(ws + OFF_U);
  float* V    = (float*)(ws + OFF_V);
  float* SQ   = (float*)(ws + OFF_SQ);
  float* EW   = (float*)(ws + OFF_EW);
  float* xcat = (float*)(ws + OFF_XCAT);
  float* xnxt = (float*)(ws + OFF_XNXT);

  k_copy_x<<<(N * FIN) / 256, 256, 0, stream>>>(x, out);

  for (int l = 0; l < NL; ++l) {
    const float* X = (l == 0) ? x : xnxt;
    const float* WkL   = Wk   + (size_t)l * H * FIN * F;
    const float* WqL   = Wq   + (size_t)l * H * FIN * F;
    const float* WqeL  = Wqe  + (size_t)l * H * (FIN + ED) * F;
    const float* aL    = a    + (size_t)l * H * 2 * F;
    const float* WoutL = Wout + (size_t)l * HF * FOUT;

    k_prep<<<12, 256, 0, stream>>>(WkL, WqL, WqeL, WoutL, aL, U, V, wqf, wqef, wof);
    k_proj<<<MT, 256, 0, stream>>>(X, wqf, wqef, hqf, hqef);
    k_sq<<<N / 256, 256, 0, stream>>>(X, U, SQ);
    k_attn<<<N, 256, 0, stream>>>(e, adj, SQ, V, attg, attng, EW);
    dim3 g3(MT, H);
    k_aggr<<<g3, 64, 0, stream>>>(attg, attng, hqf, hqef, EW, WqeL, xcat, xnxt);
    k_outp<<<MT, 256, 0, stream>>>(xcat, wof, out, l);
  }
}
